// TriaffineFull_83846351552707
// MI455X (gfx1250) — matrix operand order repaired and checked by reading, not compiled, not run on hardware
//
#include <hip/hip_runtime.h>

typedef __attribute__((ext_vector_type(16))) _Float16 v16h;
typedef __attribute__((ext_vector_type(8)))  float    v8f;

#define NI 256   // feature dim (K)
#define SB 192   // sequence length
#define NB 4     // batch

union FragH { v16h h; uint4 q[2]; };

// A operand (16x32 f16): lane provides row M=l16 (sub-tile row offset added by caller),
// elements e=0..7 -> K = half*8+e ; e=8..15 -> K = 16+half*8+(e-8). Two b128 LDS loads.
static __device__ __forceinline__ v16h load_a_frag(const _Float16* base, int stride, int row, int half) {
    FragH f;
    const _Float16* p = base + row * stride;
    f.q[0] = *(const uint4*)(p + half * 8);
    f.q[1] = *(const uint4*)(p + 16 + half * 8);
    return f.h;
}
// B operand (32x16 f16): lane provides col N=l16, elements e=0..7 -> K = half*8+e ; e=8..15 -> K = 16+half*8+(e-8).
// LDS tile stored as [n][k] row-major -> two b128 loads.
static __device__ __forceinline__ v16h load_b_frag(const _Float16* base, int stride, int n, int half) {
    FragH f;
    const _Float16* p = base + n * stride;
    f.q[0] = *(const uint4*)(p + half * 8);
    f.q[1] = *(const uint4*)(p + 16 + half * 8);
    return f.h;
}

// ---------------- Stage 0: w_red[i] = sum_{j,k} W_tri[0,i,j,k] ----------------
__global__ __launch_bounds__(256) void k_wred(const float* __restrict__ W, float* __restrict__ wred) {
    __shared__ float sd[256];
    const int i = blockIdx.x;
    const int t = threadIdx.x;
    const float4* p = (const float4*)(W + (size_t)i * (NI * NI));
    float s = 0.f;
#pragma unroll 4
    for (int j = 0; j < 64; ++j) {
        float4 v = p[(size_t)j * 256 + t];
        s += v.x + v.y + v.z + v.w;
    }
    sd[t] = s; __syncthreads();
    for (int off = 128; off > 0; off >>= 1) {
        if (t < off) sd[t] += sd[t + off];
        __syncthreads();
    }
    if (t == 0) wred[i] = sd[0];
}

// ---------------- Stage 1: hhw[b,h,i] = h_head[b,h,i] * w_red[i] ----------------
__global__ __launch_bounds__(256) void k_hhw(const float* __restrict__ hh, const float* __restrict__ wred,
                                             float* __restrict__ hhw) {
    const int idx = blockIdx.x * 256 + threadIdx.x;
    hhw[idx] = hh[idx] * wred[idx & (NI - 1)];
}

// ---------------- Generic f16-WMMA GEMM: C[MxN] = A[MxK] * B, f32 in/out ----------------
// B_NK=false: B stored row-major [K][N] (ldb = N-row stride)
// B_NK=true : B stored row-major [N][K] (i.e. B^T given; ldb = K-row stride)
// Block: 256 threads = 8 waves, tile M=64 (4 wave rows of 16) x N=64 (2 wave cols of 32), K step 32.
template <bool B_NK>
__global__ __launch_bounds__(256) void k_gemm(const float* __restrict__ A, size_t sA, int lda,
                                              const float* __restrict__ Bm, size_t sB, int ldb,
                                              float* __restrict__ C, size_t sC, int ldc) {
    __shared__ __align__(16) _Float16 Al[64 * 40];
    __shared__ __align__(16) _Float16 Bl[64 * 40];
    const int t = threadIdx.x;
    const int lane = t & 31, wave = t >> 5;
    const int l16 = lane & 15, half = lane >> 4;
    const int wm = wave >> 1, wn = wave & 1;
    const int m0 = blockIdx.x * 64, n0 = blockIdx.y * 64;
    const float* Ap = A + (size_t)blockIdx.z * sA;
    const float* Bp = Bm + (size_t)blockIdx.z * sB;
    float* Cp = C + (size_t)blockIdx.z * sC;

    v8f acc0 = {}; v8f acc1 = {};
    for (int kt = 0; kt < NI; kt += 32) {
#pragma unroll
        for (int i = 0; i < 8; ++i) {                 // A tile: 64x32
            int e = t + i * 256;
            int m = e >> 5, k = e & 31;
            Al[m * 40 + k] = (_Float16)Ap[(size_t)(m0 + m) * lda + kt + k];
        }
#pragma unroll
        for (int i = 0; i < 8; ++i) {                 // B tile -> Bl[n][k]
            int e = t + i * 256;
            if (B_NK) {
                int n = e >> 5, k = e & 31;
                Bl[n * 40 + k] = (_Float16)Bp[(size_t)(n0 + n) * ldb + kt + k];
            } else {
                int n = e & 63, k = e >> 6;
                Bl[n * 40 + k] = (_Float16)Bp[(size_t)(kt + k) * ldb + n0 + n];
            }
        }
        __syncthreads();
        v16h a  = load_a_frag(Al, 40, wm * 16 + l16, half);
        v16h b0 = load_b_frag(Bl, 40, wn * 32 + l16, half);
        v16h b1 = load_b_frag(Bl, 40, wn * 32 + 16 + l16, half);
        acc0 = __builtin_amdgcn_wmma_f32_16x16x32_f16(false, a, false, b0, (short)0, acc0, false, false);
        acc1 = __builtin_amdgcn_wmma_f32_16x16x32_f16(false, a, false, b1, (short)0, acc1, false, false);
        __syncthreads();
    }
#pragma unroll
    for (int r = 0; r < 8; ++r) {
        int M = m0 + wm * 16 + 8 * half + r;
        int Nc = n0 + wn * 32 + l16;
        Cp[(size_t)M * ldc + Nc]      = acc0[r];
        Cp[(size_t)M * ldc + Nc + 16] = acc1[r];
    }
}

// ---------------- Stage 3: fused final GEMM ----------------
// out[b,h,d,s] = sum_k (hhw[b,h,k]*h_dep[b,d,k] + HV[b,h,k] + DZ[b,d,k]) * h_sib[b,s,k]
//               + s_hd[b,h,d] + bias
// Block tile: 16 h x 16 d (M=256, m = h_loc*16 + d_loc) x 32 s; 8 waves, each 32x32 via 2x2 WMMA.
__global__ __launch_bounds__(256) void k_final(const float* __restrict__ hhw, const float* __restrict__ hdep,
                                               const float* __restrict__ HV, const float* __restrict__ DZ,
                                               const float* __restrict__ shd, const float* __restrict__ hsib,
                                               const float* __restrict__ bias, float* __restrict__ out) {
    __shared__ __align__(16) _Float16 Al[256 * 40];
    __shared__ __align__(16) _Float16 Bl[32 * 40];
    __shared__ float addl[256];
    const int t = threadIdx.x;
    const int lane = t & 31, wave = t >> 5;
    const int l16 = lane & 15, half = lane >> 4;
    const int bi = blockIdx.z;
    const int h0 = (blockIdx.y / 12) * 16;
    const int d0 = (blockIdx.y % 12) * 16;
    const int s0 = blockIdx.x * 32;
    const int hm = h0 + (t >> 4);   // this thread's A row -> h
    const int dm = d0 + (t & 15);   //                    -> d

    addl[t] = shd[((size_t)bi * SB + hm) * SB + dm] + bias[0];

    const float* hhw_p = hhw  + ((size_t)bi * SB + hm) * NI;
    const float* hd_p  = hdep + ((size_t)bi * SB + dm) * NI;
    const float* hv_p  = HV   + ((size_t)bi * SB + hm) * NI;
    const float* dz_p  = DZ   + ((size_t)bi * SB + dm) * NI;

    v8f acc[2][2] = {};
    for (int kt = 0; kt < NI; kt += 32) {
        // Build fused A row (f32 math, f16 store), row = t, 32 k values
#pragma unroll
        for (int j = 0; j < 8; ++j) {
            float4 a1 = *(const float4*)(hhw_p + kt + j * 4);
            float4 a2 = *(const float4*)(hd_p  + kt + j * 4);
            float4 a3 = *(const float4*)(hv_p  + kt + j * 4);
            float4 a4 = *(const float4*)(dz_p  + kt + j * 4);
            _Float16* dst = &Al[t * 40 + j * 4];
            dst[0] = (_Float16)(a1.x * a2.x + a3.x + a4.x);
            dst[1] = (_Float16)(a1.y * a2.y + a3.y + a4.y);
            dst[2] = (_Float16)(a1.z * a2.z + a3.z + a4.z);
            dst[3] = (_Float16)(a1.w * a2.w + a3.w + a4.w);
        }
        {   // Stage h_sib tile -> Bl[n][k], 4 elements per thread
            int e = t * 4;
            int n = e >> 5, k = e & 31;
            float4 v = *(const float4*)(hsib + ((size_t)bi * SB + s0 + n) * NI + kt + k);
            _Float16* dst = &Bl[n * 40 + k];
            dst[0] = (_Float16)v.x; dst[1] = (_Float16)v.y;
            dst[2] = (_Float16)v.z; dst[3] = (_Float16)v.w;
        }
        __syncthreads();
        v16h a0 = load_a_frag(Al, 40, wave * 32 + l16, half);
        v16h a1 = load_a_frag(Al, 40, wave * 32 + 16 + l16, half);
        v16h b0 = load_b_frag(Bl, 40, l16, half);
        v16h b1 = load_b_frag(Bl, 40, 16 + l16, half);
        acc[0][0] = __builtin_amdgcn_wmma_f32_16x16x32_f16(false, a0, false, b0, (short)0, acc[0][0], false, false);
        acc[0][1] = __builtin_amdgcn_wmma_f32_16x16x32_f16(false, a0, false, b1, (short)0, acc[0][1], false, false);
        acc[1][0] = __builtin_amdgcn_wmma_f32_16x16x32_f16(false, a1, false, b0, (short)0, acc[1][0], false, false);
        acc[1][1] = __builtin_amdgcn_wmma_f32_16x16x32_f16(false, a1, false, b1, (short)0, acc[1][1], false, false);
        __syncthreads();
    }
#pragma unroll
    for (int sm = 0; sm < 2; ++sm)
#pragma unroll
        for (int sn = 0; sn < 2; ++sn)
#pragma unroll
            for (int r = 0; r < 8; ++r) {
                int m = wave * 32 + sm * 16 + 8 * half + r;
                int h = h0 + (m >> 4), d = d0 + (m & 15);
                int s = s0 + sn * 16 + l16;
                out[(((size_t)bi * SB + h) * SB + d) * SB + s] = acc[sm][sn][r] + addl[m];
            }
}

extern "C" void kernel_launch(void* const* d_in, const int* in_sizes, int n_in,
                              void* d_out, int out_size, void* d_ws, size_t ws_size,
                              hipStream_t stream) {
    const float* h_head = (const float*)d_in[0];
    const float* h_dep  = (const float*)d_in[1];
    const float* h_sib  = (const float*)d_in[2];
    const float* W_tri  = (const float*)d_in[3];
    const float* U_hd   = (const float*)d_in[4];
    const float* V_hs   = (const float*)d_in[5];
    const float* Z_ds   = (const float*)d_in[6];
    const float* bias   = (const float*)d_in[7];
    float* out = (float*)d_out;

    float* ws   = (float*)d_ws;
    float* wred = ws;                       // 256
    float* hhw  = wred + 256;               // NB*SB*NI
    float* HU   = hhw + (size_t)NB * SB * NI;
    float* HV   = HU  + (size_t)NB * SB * NI;
    float* DZ   = HV  + (size_t)NB * SB * NI;
    float* shd  = DZ  + (size_t)NB * SB * NI;  // NB*SB*SB

    // Stage 0/1
    k_wred<<<256, 256, 0, stream>>>(W_tri, wred);
    k_hhw<<<NB * SB, 256, 0, stream>>>(h_head, wred, hhw);

    // Stage 2: HU/HV/DZ over flattened (b,h) rows: M=768, N=256, K=256; B is [K][N]
    dim3 g1(12, 4, 1);
    k_gemm<false><<<g1, 256, 0, stream>>>(h_head, 0, NI, U_hd, 0, NI, HU, 0, NI);
    k_gemm<false><<<g1, 256, 0, stream>>>(h_head, 0, NI, V_hs, 0, NI, HV, 0, NI);
    k_gemm<false><<<g1, 256, 0, stream>>>(h_dep,  0, NI, Z_ds, 0, NI, DZ, 0, NI);

    // s_hd[b] = HU[b] (192x256) * h_dep[b]^T : B given as [N][K], batched over b
    dim3 g2(3, 3, NB);
    k_gemm<true><<<g2, 256, 0, stream>>>(HU, (size_t)SB * NI, NI,
                                         h_dep, (size_t)SB * NI, NI,
                                         shd, (size_t)SB * SB, SB);

    // Stage 3: fused trilinear + biaffine GEMM -> output
    dim3 g3(6, 144, NB);
    k_final<<<g3, 256, 0, stream>>>(hhw, h_dep, HV, DZ, shd, h_sib, bias, out);
}
